// SlidingWindowAttention_72284299592329
// MI455X (gfx1250) — matrix operand order repaired and checked by reading, not compiled, not run on hardware
//
#include <hip/hip_runtime.h>

typedef __bf16 bf16;
typedef __attribute__((ext_vector_type(16))) __bf16 v16bf;
typedef __attribute__((ext_vector_type(8)))  float  v8f;
typedef int aiv4 __attribute__((vector_size(16)));   // matches builtin's V4i pointee

#define S_LEN  4096
#define DMODEL 1024
#define NHEAD  16
#define HDIM   64
#define PADL   256
#define SPAD   4672   // 256 left pad + 4096 + 320 right pad

// ---- workspace layout (bytes) ----
#define OFF_XB    ((size_t)0)                         // x bf16        [4096][1024]  8 MB
#define OFF_WQKVB ((size_t)8388608)                   // Wqkv bf16     [3072][1024]  6 MB
#define OFF_WOUTB ((size_t)14680064)                  // Wout bf16     [1024][1024]  2 MB
#define OFF_QB    ((size_t)16777216)                  // Q bf16        [16][4096][64]
#define OFF_KB    ((size_t)25165824)                  // K bf16 padded [16][4672][64]
#define OFF_VT    ((size_t)34734080)                  // V^T bf16 pad  [16][64][4672]
#define OFF_OB    ((size_t)44302336)                  // attn out bf16 [4096][1024]
#define KB_BYTES  ((size_t)NHEAD * SPAD * HDIM * 2)
#define VT_BYTES  ((size_t)NHEAD * HDIM * SPAD * 2)

union U16 { v16bf v; uint4 q[2]; };

// ---- async global->LDS copy (gfx1250), with safe fallback ----
#if defined(__has_builtin)
#if __has_builtin(__builtin_amdgcn_global_load_async_to_lds_b128)
#define HAVE_ASYNC_LDS 1
#endif
#endif

#ifdef HAVE_ASYNC_LDS
#define ASYNC_CP16(gp, lp)                                                 \
  __builtin_amdgcn_global_load_async_to_lds_b128(                          \
      (__attribute__((address_space(1))) aiv4*)(gp),                       \
      (__attribute__((address_space(3))) aiv4*)(lp), 0, 0)
#if __has_builtin(__builtin_amdgcn_s_wait_asynccnt)
#define ASYNC_WAIT(n) __builtin_amdgcn_s_wait_asynccnt(n)
#else
#define ASYNC_WAIT(n) asm volatile("s_wait_asynccnt %0" ::"i"(n) : "memory")
#endif
#else
#define ASYNC_CP16(gp, lp) (*(uint4*)(lp) = *(const uint4*)(gp))
#define ASYNC_WAIT(n) ((void)0)
#endif

__device__ __forceinline__ v8f wmma_bf16(v16bf a, v16bf b, v8f c) {
  return __builtin_amdgcn_wmma_f32_16x16x32_bf16(false, a, false, b, (short)0, c,
                                                 false, false);
}

__device__ __forceinline__ float rowmax16(float v) {
  v = fmaxf(v, __shfl_xor(v, 1));
  v = fmaxf(v, __shfl_xor(v, 2));
  v = fmaxf(v, __shfl_xor(v, 4));
  v = fmaxf(v, __shfl_xor(v, 8));
  return v;
}
__device__ __forceinline__ float rowsum16(float v) {
  v += __shfl_xor(v, 1);
  v += __shfl_xor(v, 2);
  v += __shfl_xor(v, 4);
  v += __shfl_xor(v, 8);
  return v;
}

// ---------------- fp32 -> bf16 conversion (vector x4) ----------------
__global__ void cvt_f32_bf16_x4(const float* __restrict__ src,
                                bf16* __restrict__ dst, int n4) {
  int i = blockIdx.x * blockDim.x + threadIdx.x;
  if (i < n4) {
    float4 f = ((const float4*)src)[i];
    union { bf16 b[4]; uint2 u; } t;
    t.b[0] = (bf16)f.x; t.b[1] = (bf16)f.y; t.b[2] = (bf16)f.z; t.b[3] = (bf16)f.w;
    ((uint2*)dst)[i] = t.u;
  }
}

// ---------------- LDS-staged 128x64 GEMM core ----------------
// Block (256 thr, 8 waves) computes rows [0,128) of Ag x cols [0,64) of Bg^T
// (both row-major with K contiguous, K = DMODEL).  A/B k-slices are staged
// into double-buffered LDS via async global->LDS DMA; WMMA reads from LDS.
__device__ __forceinline__ void gemm128x64(const bf16* __restrict__ Ag,
                                           const bf16* __restrict__ Bg,
                                           bf16* Asm, bf16* Bsm, v8f c[4]) {
  const int t = threadIdx.x;
  const int lane = t & 31, w = t >> 5;
  const int lo = lane & 15, hi = lane >> 4;
  const int r4 = t >> 2;          // 0..63
  const int o8 = (t & 3) * 8;     // 0,8,16,24 halves

  auto stage = [&](int buf, int kk) {
    bf16* Ad = Asm + buf * (128 * 32);
    bf16* Bd = Bsm + buf * (64 * 32);
    ASYNC_CP16(Ag + (size_t)r4 * DMODEL + kk + o8,        Ad + r4 * 32 + o8);
    ASYNC_CP16(Ag + (size_t)(64 + r4) * DMODEL + kk + o8, Ad + (64 + r4) * 32 + o8);
    ASYNC_CP16(Bg + (size_t)r4 * DMODEL + kk + o8,        Bd + r4 * 32 + o8);
    __builtin_prefetch(Ag + (size_t)r4 * DMODEL + kk + o8 + 64, 0, 1);
    __builtin_prefetch(Bg + (size_t)r4 * DMODEL + kk + o8 + 64, 0, 1);
  };

  stage(0, 0);
  for (int it = 0; it < DMODEL / 32; it++) {
    const int b = it & 1;
    if (it + 1 < DMODEL / 32) {
      stage(1 - b, (it + 1) * 32);   // prev compute finished at trailing barrier
      ASYNC_WAIT(3);                 // 3 in-flight per wave = the stage just issued
    } else {
      ASYNC_WAIT(0);
    }
    __syncthreads();                 // staged data visible to all waves

    const bf16* Al = Asm + b * (128 * 32) + (w * 16 + lo) * 32;
    U16 a;
    a.q[0] = *(const uint4*)(Al + hi * 8);
    a.q[1] = *(const uint4*)(Al + 16 + hi * 8);
    const bf16* Bl = Bsm + b * (64 * 32);
#pragma unroll
    for (int tt = 0; tt < 4; tt++) {
      const bf16* bp = Bl + (tt * 16 + lo) * 32 + hi * 8;
      U16 bb;
      bb.q[0] = *(const uint4*)(bp);
      bb.q[1] = *(const uint4*)(bp + 16);
      c[tt] = wmma_bf16(a.v, bb.v, c[tt]);
    }
    __syncthreads();                 // all reads done before buffer reuse
  }
}

// ---------------- QKV projection GEMM ----------------
// grid (32, 48), block 256.
__global__ __launch_bounds__(256) void qkv_gemm(
    const bf16* __restrict__ Xb, const bf16* __restrict__ Wb,
    const float* __restrict__ bqkv,
    bf16* __restrict__ Qb, bf16* __restrict__ Kb, bf16* __restrict__ Vt) {
  __shared__ bf16 Asm[2][128 * 32];
  __shared__ bf16 Bsm[2][64 * 32];
  const int mb = blockIdx.x;
  const int col0 = blockIdx.y * 64;

  v8f c[4] = {};
  gemm128x64(Xb + (size_t)mb * 128 * DMODEL, Wb + (size_t)col0 * DMODEL,
             &Asm[0][0], &Bsm[0][0], c);

  const int lane = threadIdx.x & 31, w = threadIdx.x >> 5;
  const int lo = lane & 15, hi = lane >> 4;
  for (int tt = 0; tt < 4; tt++) {
    for (int r = 0; r < 8; r++) {
      int srow = mb * 128 + w * 16 + r + 8 * hi;
      int col  = col0 + tt * 16 + lo;
      float acc = c[tt][r] + bqkv[col];
      bf16 val = (bf16)acc;
      int tsel = col >> 10;
      int hh   = (col & 1023) >> 6;
      int dd   = col & 63;
      if (tsel == 0)       Qb[((size_t)hh * S_LEN + srow) * HDIM + dd] = val;
      else if (tsel == 1)  Kb[((size_t)hh * SPAD + srow + PADL) * HDIM + dd] = val;
      else                 Vt[((size_t)hh * HDIM + dd) * SPAD + srow + PADL] = val;
    }
  }
}

// ---------------- sliding-window attention ----------------
// grid 512, block 256 (8 waves). One wave per (head, 16-query block).
__global__ __launch_bounds__(256) void attn_kernel(
    const bf16* __restrict__ Qb, const bf16* __restrict__ Kb,
    const bf16* __restrict__ Vt, bf16* __restrict__ Ob) {
  __shared__ bf16 plds[8][16][32];   // wave-private P tiles (C/D -> A layout bounce)

  const int lane = threadIdx.x & 31;
  const int w    = threadIdx.x >> 5;
  const int lo = lane & 15, hi = lane >> 4;
  const int gw = blockIdx.x * 8 + w;
  const int h  = gw >> 8;          // 256 query blocks per head
  const int q0 = (gw & 255) * 16;

  // Q tile, 16x64 bf16, as two A operands (d=0..31, d=32..63)
  U16 aq0, aq1;
  const bf16* qrow = Qb + ((size_t)h * S_LEN + q0 + lo) * HDIM;
  aq0.q[0] = *(const uint4*)(qrow + hi * 8);
  aq0.q[1] = *(const uint4*)(qrow + 16 + hi * 8);
  aq1.q[0] = *(const uint4*)(qrow + 32 + hi * 8);
  aq1.q[1] = *(const uint4*)(qrow + 48 + hi * 8);

  v8f o0 = {}, o1 = {}, o2 = {}, o3 = {};
  float m_i[8], l_i[8];
#pragma unroll
  for (int r = 0; r < 8; r++) { m_i[r] = -1e30f; l_i[r] = 0.0f; }

  for (int j = 0; j < 17; j++) {
    const int kt = q0 - 256 + j * 32;

    // ---- scores: two 16x16 subtiles over 32 keys ----
    v8f s01[2];
#pragma unroll
    for (int s2 = 0; s2 < 2; s2++) {
      int key = kt + s2 * 16 + lo;                 // one key per lane (padded buffer)
      const bf16* krow = Kb + ((size_t)h * SPAD + key + PADL) * HDIM;
      U16 bk0, bk1;
      bk0.q[0] = *(const uint4*)(krow + hi * 8);
      bk0.q[1] = *(const uint4*)(krow + hi * 8 + 16);
      bk1.q[0] = *(const uint4*)(krow + 32 + hi * 8);
      bk1.q[1] = *(const uint4*)(krow + 32 + hi * 8 + 16);
      __builtin_prefetch(krow + 32 * HDIM, 0, 1);
      v8f sc = {};
      sc = wmma_bf16(aq0.v, bk0.v, sc);
      sc = wmma_bf16(aq1.v, bk1.v, sc);
      s01[s2] = sc;
    }

    // ---- mask + online softmax ----
#pragma unroll
    for (int r = 0; r < 8; r++) {
      int dq = j * 32 + lo - (r + 8 * hi);         // keypos - (qpos - 256)
      float sA = (dq >= 0 && dq < 512)           ? s01[0][r] * 0.125f : -1e30f;
      float sB = (dq + 16 >= 0 && dq + 16 < 512) ? s01[1][r] * 0.125f : -1e30f;
      float tm   = rowmax16(fmaxf(sA, sB));
      float mnew = fmaxf(m_i[r], tm);
      float corr = __expf(m_i[r] - mnew);
      float pa = __expf(sA - mnew);
      float pb = __expf(sB - mnew);
      l_i[r] = l_i[r] * corr + rowsum16(pa + pb);
      m_i[r] = mnew;
      o0[r] *= corr; o1[r] *= corr; o2[r] *= corr; o3[r] *= corr;
      plds[w][r + 8 * hi][lo]      = (bf16)pa;
      plds[w][r + 8 * hi][16 + lo] = (bf16)pb;
    }
    __builtin_amdgcn_wave_barrier();

    // ---- P (A layout) from LDS ----
    U16 pa16;
    const bf16* prow = &plds[w][lo][0];
    pa16.q[0] = *(const uint4*)(prow + hi * 8);
    pa16.q[1] = *(const uint4*)(prow + 16 + hi * 8);

    // ---- o += P @ V ; V^T is K-contiguous per d-column ----
    {
      const bf16* vb = Vt + (size_t)(h * HDIM + lo) * SPAD + (kt + PADL) + hi * 8;
      U16 bv;
      bv.q[0] = *(const uint4*)(vb);
      bv.q[1] = *(const uint4*)(vb + 16);
      o0 = wmma_bf16(pa16.v, bv.v, o0);
      bv.q[0] = *(const uint4*)(vb + 16 * SPAD);
      bv.q[1] = *(const uint4*)(vb + 16 * SPAD + 16);
      o1 = wmma_bf16(pa16.v, bv.v, o1);
      bv.q[0] = *(const uint4*)(vb + 32 * SPAD);
      bv.q[1] = *(const uint4*)(vb + 32 * SPAD + 16);
      o2 = wmma_bf16(pa16.v, bv.v, o2);
      bv.q[0] = *(const uint4*)(vb + 48 * SPAD);
      bv.q[1] = *(const uint4*)(vb + 48 * SPAD + 16);
      o3 = wmma_bf16(pa16.v, bv.v, o3);
    }
  }

  // ---- finalize: divide by l, store bf16 [S][D] row-major ----
#pragma unroll
  for (int r = 0; r < 8; r++) {
    float inv = 1.0f / l_i[r];
    int srow = q0 + r + 8 * hi;
    bf16* orow = Ob + (size_t)srow * DMODEL + h * HDIM;
    orow[lo]      = (bf16)(o0[r] * inv);
    orow[16 + lo] = (bf16)(o1[r] * inv);
    orow[32 + lo] = (bf16)(o2[r] * inv);
    orow[48 + lo] = (bf16)(o3[r] * inv);
  }
}

// ---------------- output projection GEMM ----------------
// grid (32, 16), block 256. fp32 output.
__global__ __launch_bounds__(256) void out_gemm(
    const bf16* __restrict__ Ob, const bf16* __restrict__ Wb,
    const float* __restrict__ bout, float* __restrict__ out) {
  __shared__ bf16 Asm[2][128 * 32];
  __shared__ bf16 Bsm[2][64 * 32];
  const int mb = blockIdx.x;
  const int col0 = blockIdx.y * 64;

  v8f c[4] = {};
  gemm128x64(Ob + (size_t)mb * 128 * DMODEL, Wb + (size_t)col0 * DMODEL,
             &Asm[0][0], &Bsm[0][0], c);

  const int lane = threadIdx.x & 31, w = threadIdx.x >> 5;
  const int lo = lane & 15, hi = lane >> 4;
  for (int tt = 0; tt < 4; tt++) {
    for (int r = 0; r < 8; r++) {
      int srow = mb * 128 + w * 16 + r + 8 * hi;
      int col  = col0 + tt * 16 + lo;
      out[(size_t)srow * DMODEL + col] = c[tt][r] + bout[col];
    }
  }
}

extern "C" void kernel_launch(void* const* d_in, const int* in_sizes, int n_in,
                              void* d_out, int out_size, void* d_ws, size_t ws_size,
                              hipStream_t stream) {
  const float* x    = (const float*)d_in[0];
  const float* Wqkv = (const float*)d_in[1];
  const float* bqkv = (const float*)d_in[2];
  const float* Wout = (const float*)d_in[3];
  const float* bout = (const float*)d_in[4];
  float* out = (float*)d_out;

  char* ws = (char*)d_ws;
  bf16* Xb    = (bf16*)(ws + OFF_XB);
  bf16* Wqkvb = (bf16*)(ws + OFF_WQKVB);
  bf16* Woutb = (bf16*)(ws + OFF_WOUTB);
  bf16* Qb    = (bf16*)(ws + OFF_QB);
  bf16* Kb    = (bf16*)(ws + OFF_KB);
  bf16* Vt    = (bf16*)(ws + OFF_VT);
  bf16* Ob    = (bf16*)(ws + OFF_OB);

  // zero padded K / V^T buffers (out-of-sequence keys must read as 0)
  (void)hipMemsetAsync(Kb, 0, KB_BYTES, stream);
  (void)hipMemsetAsync(Vt, 0, VT_BYTES, stream);

  // fp32 -> bf16 conversions
  cvt_f32_bf16_x4<<<(S_LEN * DMODEL / 4 + 255) / 256, 256, 0, stream>>>(x, Xb, S_LEN * DMODEL / 4);
  cvt_f32_bf16_x4<<<(3 * DMODEL * DMODEL / 4 + 255) / 256, 256, 0, stream>>>(Wqkv, Wqkvb, 3 * DMODEL * DMODEL / 4);
  cvt_f32_bf16_x4<<<(DMODEL * DMODEL / 4 + 255) / 256, 256, 0, stream>>>(Wout, Woutb, DMODEL * DMODEL / 4);

  // QKV projection, attention, output projection
  qkv_gemm<<<dim3(32, 48), 256, 0, stream>>>(Xb, Wqkvb, bqkv, Qb, Kb, Vt);
  attn_kernel<<<512, 256, 0, stream>>>(Qb, Kb, Vt, Ob);
  out_gemm<<<dim3(32, 16), 256, 0, stream>>>(Ob, Woutb, bout, out);
}
